// MoeLayer_17703855194815
// MI455X (gfx1250) — matrix operand order repaired and checked by reading, not compiled, not run on hardware
//
#include <hip/hip_runtime.h>
#include <hip/hip_bf16.h>

// out = X @ W0^T  (router argmax over a size-1 axis is always expert 0)
// X: [N, 256] f32, W0: [256, 256] f32 (expert 0 of w_experts), out: [N, 256] f32
//
// bf16x3 split-precision GEMM on v_wmma_f32_16x16x32_bf16:
//   x = x_hi + x_lo, w = w_hi + w_lo (bf16 RNE splits)
//   x@w ~= x_hi@w_hi + x_hi@w_lo + x_lo@w_hi   (fp32 accumulate)
// W0 hi/lo tiles live entirely in LDS (264 KB of 320 KB WGP LDS); X streamed
// from HBM once with NT hints -> ~23 us HBM roofline. Four N-tiles per group
// with phase-ordered WMMAs put distance 4 between same-accumulator WMMAs,
// covering the 1-NOP+4-coexec BF16 WMMA->WMMA hazard window.

typedef __attribute__((ext_vector_type(16))) __bf16 v16bf;
typedef __attribute__((ext_vector_type(8)))  float  v8f;
typedef __attribute__((ext_vector_type(4)))  float  v4f;   // native vector for NT loads

#define D 256
#define ROW_BYTES 528              // 256 bf16 (512B) + 16B pad: conflict-free ds_load_b128
#define W_LO_OFF (256 * ROW_BYTES) // 135168
#define LDS_BYTES (2 * 256 * ROW_BYTES) // 270336 <= 320KB

union V16 {
    v16bf v;
    unsigned int   u[8];
    unsigned short s[16];
};

__device__ __forceinline__ unsigned short f2bf(float f) {
    unsigned int u = __float_as_uint(f);
    unsigned int r = u + 0x7FFFu + ((u >> 16) & 1u); // round-to-nearest-even
    return (unsigned short)(r >> 16);
}
__device__ __forceinline__ float bf2f(unsigned short h) {
    return __uint_as_float(((unsigned int)h) << 16);
}

__global__ __launch_bounds__(256) void moe_gemm_bf16x3_kernel(
    const float* __restrict__ X,    // [N, 256]
    const float* __restrict__ W0,   // [256, 256] expert 0, row-major [out, in]
    float* __restrict__ OUT,        // [N, 256]
    int N)
{
    extern __shared__ char lds[];

    // ---- Stage W0 into LDS as bf16 hi/lo, row-major with padded rows ----
    {
        const int row = threadIdx.x;            // 256 threads, one W row each
        const float* wr = W0 + row * D;
        unsigned int* hi_row = (unsigned int*)(lds + row * ROW_BYTES);
        unsigned int* lo_row = (unsigned int*)(lds + W_LO_OFF + row * ROW_BYTES);
        #pragma unroll 4
        for (int k = 0; k < D; k += 2) {
            float f0 = wr[k], f1 = wr[k + 1];
            unsigned short h0 = f2bf(f0), h1 = f2bf(f1);
            unsigned short l0 = f2bf(f0 - bf2f(h0));
            unsigned short l1 = f2bf(f1 - bf2f(h1));
            hi_row[k >> 1] = (unsigned int)h0 | ((unsigned int)h1 << 16);
            lo_row[k >> 1] = (unsigned int)l0 | ((unsigned int)l1 << 16);
        }
    }
    __syncthreads();

    // ---- Per-wave 16-row strip, full 256 output columns ----
    const int lane = threadIdx.x & 31;
    const int wave = threadIdx.x >> 5;
    const int l15  = lane & 15;
    const int hv   = lane >> 4;   // half-wave selector (A/B K swizzle)

    const long n0 = (long)blockIdx.x * 128 + (long)wave * 16;
    const float* xrow = X + (n0 + l15) * D;
    const char* bbase = lds + l15 * ROW_BYTES + hv * 16; // per-lane B column base

    v8f acc[16];
    #pragma unroll
    for (int t = 0; t < 16; ++t)
        #pragma unroll
        for (int j = 0; j < 8; ++j)
            acc[t][j] = 0.0f;

    #pragma unroll 1
    for (int ks = 0; ks < 8; ++ks) {
        const int k0 = ks * 32;
        const int kA = k0 + hv * 8;

        // A fragment per ISA 16-bit A 16x32 layout:
        // lanes 0-15: elems 0..7 = K k0+0..7,  elems 8..15 = K k0+16..23
        // lanes16-31: elems 0..7 = K k0+8..15, elems 8..15 = K k0+24..31
        // X is consumed exactly once -> non-temporal loads keep L2 for W0.
        v4f fa0 = __builtin_nontemporal_load((const v4f*)(xrow + kA));
        v4f fa1 = __builtin_nontemporal_load((const v4f*)(xrow + kA + 4));
        v4f fb0 = __builtin_nontemporal_load((const v4f*)(xrow + kA + 16));
        v4f fb1 = __builtin_nontemporal_load((const v4f*)(xrow + kA + 20));

        float fv[16] = { fa0[0], fa0[1], fa0[2], fa0[3],
                         fa1[0], fa1[1], fa1[2], fa1[3],
                         fb0[0], fb0[1], fb0[2], fb0[3],
                         fb1[0], fb1[1], fb1[2], fb1[3] };
        V16 ahi, alo;
        #pragma unroll
        for (int e = 0; e < 16; ++e) {
            unsigned short h = f2bf(fv[e]);
            ahi.s[e] = h;
            alo.s[e] = f2bf(fv[e] - bf2f(h));
        }

        const char* bk = bbase + k0 * 2;

        // Four tiles per group, phase-ordered WMMAs:
        // same-accumulator dependency distance = 4 (covers hazard window).
        #pragma unroll
        for (int t = 0; t < 16; t += 4) {
            V16 bh[4], bl[4];
            #pragma unroll
            for (int q = 0; q < 4; ++q) {
                const char* p = bk + (t + q) * (16 * ROW_BYTES);
                *(uint4*)&bh[q].u[0] = *(const uint4*)(p);
                *(uint4*)&bh[q].u[4] = *(const uint4*)(p + 32);
                *(uint4*)&bl[q].u[0] = *(const uint4*)(p + W_LO_OFF);
                *(uint4*)&bl[q].u[4] = *(const uint4*)(p + W_LO_OFF + 32);
            }
            #pragma unroll
            for (int q = 0; q < 4; ++q)
                acc[t + q] = __builtin_amdgcn_wmma_f32_16x16x32_bf16(
                    false, ahi.v, false, bh[q].v, (short)0, acc[t + q], false, false);
            #pragma unroll
            for (int q = 0; q < 4; ++q)
                acc[t + q] = __builtin_amdgcn_wmma_f32_16x16x32_bf16(
                    false, ahi.v, false, bl[q].v, (short)0, acc[t + q], false, false);
            #pragma unroll
            for (int q = 0; q < 4; ++q)
                acc[t + q] = __builtin_amdgcn_wmma_f32_16x16x32_bf16(
                    false, alo.v, false, bh[q].v, (short)0, acc[t + q], false, false);
        }
    }

    // ---- Store: C/D layout -> VGPR j is row (j + hv*8), lane l15 is column ----
    // Output is written exactly once -> non-temporal stores.
    #pragma unroll
    for (int t = 0; t < 16; ++t) {
        #pragma unroll
        for (int j = 0; j < 8; ++j) {
            __builtin_nontemporal_store(
                acc[t][j], &OUT[(n0 + j + hv * 8) * D + t * 16 + l15]);
        }
    }
}

extern "C" void kernel_launch(void* const* d_in, const int* in_sizes, int n_in,
                              void* d_out, int out_size, void* d_ws, size_t ws_size,
                              hipStream_t stream) {
    const float* X  = (const float*)d_in[0];   // inputs  [N, 256]
    const float* We = (const float*)d_in[1];   // w_experts [8, 256, 256]; expert 0 = first 65536
    // d_in[2] = w_router, structurally unused (argmax over size-1 axis == 0)
    float* OUT = (float*)d_out;

    const int N = in_sizes[0] / D;             // 262144
    const int blocks = N / 128;                // 128 rows per workgroup (8 waves x 16 rows)

    (void)hipFuncSetAttribute((const void*)moe_gemm_bf16x3_kernel,
                              hipFuncAttributeMaxDynamicSharedMemorySize, LDS_BYTES);

    moe_gemm_bf16x3_kernel<<<blocks, 256, LDS_BYTES, stream>>>(X, We, OUT, N);
}
